// QuantizeChannelWise_39041252720883
// MI455X (gfx1250) — matrix operand order repaired and checked by reading, not compiled, not run on hardware
//
#include <hip/hip_runtime.h>

typedef __bf16 bf16;
typedef __attribute__((ext_vector_type(16))) __bf16 v16bf;
typedef __attribute__((ext_vector_type(8)))  __bf16 v8bf;
typedef __attribute__((ext_vector_type(8)))  float  v8f;

#define NQ      8192   // N*C queries
#define DDIM    1024   // H*W
#define KCB     1024   // codebook entries
#define XSTRIDE 1032   // LDS row stride in bf16 (1024 + 8 pad -> 4-bank shift/row)

#define WAIT_ASYNC0() asm volatile("s_wait_asynccnt 0x0" ::: "memory")
#define WAIT_DS0()    asm volatile("s_wait_dscnt 0x0" ::: "memory")

// ---------------------------------------------------------------------------
// Prep: codebook f32 -> bf16 (for WMMA) + per-codeword squared norm (f32).
// ---------------------------------------------------------------------------
__global__ __launch_bounds__(256) void vq_prep(const float* __restrict__ cb,
                                               bf16* __restrict__ cbb,
                                               float* __restrict__ cnorm) {
  const int k   = blockIdx.x;
  const int tid = threadIdx.x;

  const float4 f = ((const float4*)(cb + (size_t)k * DDIM))[tid];
  bf16* d = cbb + (size_t)k * DDIM + tid * 4;
  d[0] = (bf16)f.x; d[1] = (bf16)f.y; d[2] = (bf16)f.z; d[3] = (bf16)f.w;

  float s = f.x * f.x + f.y * f.y + f.z * f.z + f.w * f.w;
#pragma unroll
  for (int off = 16; off >= 1; off >>= 1) s += __shfl_xor(s, off, 32);

  __shared__ float red[8];
  if ((tid & 31) == 0) red[tid >> 5] = s;
  __syncthreads();
  if (tid == 0) {
    float t = 0.f;
#pragma unroll
    for (int w = 0; w < 8; ++w) t += red[w];
    cnorm[k] = t;
  }
}

// ---------------------------------------------------------------------------
// Async-stage one 16x32 bf16 codebook chunk (1 KB) into a per-wave LDS buffer.
// Chunk layout: row-major [16 codewords][32 K] bf16. Two B128 async ops:
// lanes 0..31 copy 16B each -> bytes [0,512) = rows 0..7, [512,1024) = rows 8..15.
// ---------------------------------------------------------------------------
__device__ __forceinline__ void async_b_chunk(const bf16* __restrict__ cbb,
                                              int k0, int kb, unsigned lds_dst,
                                              int lane) {
  const int row = lane >> 2;            // 0..7
  const int col = (lane & 3) * 16;      // byte offset within 64B row
  const char* g0 = (const char*)cbb + ((size_t)(k0 + row)     * DDIM) * 2 + kb * 2 + col;
  const char* g1 = (const char*)cbb + ((size_t)(k0 + 8 + row) * DDIM) * 2 + kb * 2 + col;
  const unsigned d0 = lds_dst + lane * 16;
  const unsigned d1 = lds_dst + 512 + lane * 16;
  asm volatile("global_load_async_to_lds_b128 %0, %1, off" :: "v"(d0), "v"(g0) : "memory");
  asm volatile("global_load_async_to_lds_b128 %0, %1, off" :: "v"(d1), "v"(g1) : "memory");
}

// One pipeline phase: load A/B fragments (DS), fence DS, issue next-chunk
// prefetch into the other buffer, then WMMA (async loads fly under the MMA).
__device__ __forceinline__ void phase(v8f& acc,
                                      const bf16* __restrict__ arow, int kb,
                                      const bf16* __restrict__ bbuf, int n, int hf,
                                      const bf16* __restrict__ cbb, int k0, int kb_next,
                                      unsigned lds_next, int lane, bool do_pref) {
  union { v16bf v; v8bf h[2]; } A, B;
  A.h[0] = *(const v8bf*)(arow + kb);        // K = kb + 8*hf + 0..7
  A.h[1] = *(const v8bf*)(arow + kb + 16);   // K = kb + 16 + 8*hf + 0..7
  const bf16* bs = bbuf + n * 32 + hf * 8;   // row n, K = 8*hf+0..7, 16+8*hf+0..7
  B.h[0] = *(const v8bf*)(bs);
  B.h[1] = *(const v8bf*)(bs + 16);
  WAIT_DS0();                                // reads landed -> safe to overwrite other buf
  if (do_pref) async_b_chunk(cbb, k0, kb_next, lds_next, lane);
  acc = __builtin_amdgcn_wmma_f32_16x16x32_bf16(
      false, A.v, false, B.v, (short)0, acc, false, false);
}

// ---------------------------------------------------------------------------
// Main: per 16-query tile, bf16 WMMA Gram vs all 1024 codewords with async
// double-buffered codebook staging, fused argmin of (cnorm[k] - 2*dot),
// then fp32 gather of selected codewords to both output halves.
// ---------------------------------------------------------------------------
__global__ __launch_bounds__(256) void vq_main(const float* __restrict__ x,
                                               const float* __restrict__ cb,
                                               const bf16* __restrict__ cbb,
                                               const float* __restrict__ cnorm,
                                               float* __restrict__ out) {
  __shared__ __align__(16) bf16 ldsx[16 * XSTRIDE];
  __shared__ __align__(16) bf16 bstage[8 * 1024];  // 8 waves x 2 bufs x 512 bf16
  __shared__ float wbd[8][16];
  __shared__ int   wbi[8][16];
  __shared__ int   sidx[16];

  const int tid = threadIdx.x;
  const int m0  = blockIdx.x * 16;

  // Stage 16x1024 f32 x-tile into LDS as bf16 (coalesced float4 reads).
  for (int v = tid; v < 16 * 256; v += 256) {
    const int m = v >> 8, c4 = v & 255;
    const float4 f = ((const float4*)x)[(size_t)(m0 + m) * 256 + c4];
    bf16* d = &ldsx[m * XSTRIDE + c4 * 4];
    d[0] = (bf16)f.x; d[1] = (bf16)f.y; d[2] = (bf16)f.z; d[3] = (bf16)f.w;
  }
  __syncthreads();

  const int lane = tid & 31;
  const int wv   = tid >> 5;      // 8 waves per block
  const int hf   = lane >> 4;     // lane half: K-phase of A/B frags
  const int n    = lane & 15;     // A: row M; B/C: column N

  bf16*          wbuf = &bstage[wv * 1024];            // buf0 = [0,512), buf1 = [512,1024)
  const unsigned wlds = (unsigned)(uintptr_t)wbuf;     // LDS byte address (low 32 bits)

  float bd[8];
  int   bi[8];
#pragma unroll
  for (int r = 0; r < 8; ++r) { bd[r] = 3.4e38f; bi[r] = 0; }

  // Each wave owns codeword tiles t = wv, wv+8, ... (8 tiles of 16 codewords).
  for (int t = wv; t < KCB / 16; t += 8) {
    const int k0 = t * 16;
    v8f acc = {};
    const bf16* arow = &ldsx[n * XSTRIDE + hf * 8];
    const bf16* buf0 = wbuf;
    const bf16* buf1 = wbuf + 512;

    async_b_chunk(cbb, k0, 0, wlds, lane);             // prime buf0

    for (int kb = 0; kb < DDIM; kb += 64) {
      WAIT_ASYNC0();                                   // buf0 chunk ready
      phase(acc, arow, kb, buf0, n, hf,
            cbb, k0, kb + 32, wlds + 1024, lane, true);
      WAIT_ASYNC0();                                   // buf1 chunk ready
      phase(acc, arow, kb + 32, buf1, n, hf,
            cbb, k0, kb + 64, wlds, lane, kb + 64 < DDIM);
    }

    // C layout: acc[r] = dot(x_{M=r+8*hf}, c_{k0+n}); ||x||^2 drops from argmin.
    const float cn = cnorm[k0 + n];
#pragma unroll
    for (int r = 0; r < 8; ++r) {
      const float dist = cn - 2.0f * acc[r];
      if (dist < bd[r]) { bd[r] = dist; bi[r] = k0 + n; }
    }
  }

  // Cross-lane argmin over N (width-16 keeps the two M-halves separate);
  // tie-break to lowest index to match jnp.argmin (first occurrence).
#pragma unroll
  for (int r = 0; r < 8; ++r) {
    float d = bd[r];
    int   i = bi[r];
#pragma unroll
    for (int off = 8; off >= 1; off >>= 1) {
      const float od = __shfl_xor(d, off, 16);
      const int   oi = __shfl_xor(i, off, 16);
      if (od < d || (od == d && oi < i)) { d = od; i = oi; }
    }
    if (n == 0) { wbd[wv][r + 8 * hf] = d; wbi[wv][r + 8 * hf] = i; }
  }
  __syncthreads();

  // Cross-wave argmin -> final index per query row.
  if (tid < 16) {
    float d = wbd[0][tid];
    int   i = wbi[0][tid];
#pragma unroll
    for (int w = 1; w < 8; ++w) {
      const float od = wbd[w][tid];
      const int   oi = wbi[w][tid];
      if (od < d || (od == d && oi < i)) { d = od; i = oi; }
    }
    sidx[tid] = i;
  }
  __syncthreads();

  // Gather selected fp32 codewords; write both output halves (out, selected).
  float4*       o4  = (float4*)out;
  const float4* c4p = (const float4*)cb;
  const size_t  second = (size_t)NQ * DDIM / 4;  // float4 units
  for (int v = tid; v < 16 * 256; v += 256) {
    const int m = v >> 8, c4 = v & 255;
    const float4 val = c4p[(size_t)sidx[m] * 256 + c4];
    const size_t o = (size_t)(m0 + m) * 256 + c4;
    o4[o]          = val;
    o4[o + second] = val;
  }
}

extern "C" void kernel_launch(void* const* d_in, const int* in_sizes, int n_in,
                              void* d_out, int out_size, void* d_ws, size_t ws_size,
                              hipStream_t stream) {
  (void)in_sizes; (void)n_in; (void)out_size; (void)ws_size;
  const float* x  = (const float*)d_in[0];   // (64,128,32,32) f32
  const float* cb = (const float*)d_in[1];   // (1024,1,32,32) f32

  bf16*  cbb   = (bf16*)d_ws;                                               // 2 MB
  float* cnorm = (float*)((char*)d_ws + (size_t)KCB * DDIM * sizeof(bf16)); // 4 KB

  vq_prep<<<KCB, 256, 0, stream>>>(cb, cbb, cnorm);
  vq_main<<<NQ / 16, 256, 0, stream>>>(x, cb, cbb, cnorm, (float*)d_out);
}
